// TwoWLConv_90924457656371
// MI455X (gfx1250) — 8-bit matrix operand order repaired and checked by reading, compiled, not run on hardware
//
#include <hip/hip_runtime.h>
#include <hip/hip_bf16.h>
#include <math.h>


typedef int v8i __attribute__((ext_vector_type(8)));

#define NUM_COLORS 16
#define EMPTY_KEY 0xFFFFFFFFu
#define CHUNK 4096

__device__ __forceinline__ unsigned long long onehot8(int val, int colorBase)
{
    unsigned rel = (unsigned)(val - colorBase);
    return (rel < 8u) ? (1ull << (rel * 8u)) : 0ull;
}

template <bool ROWM>
__global__ __launch_bounds__(32)
void hist16_wmma_kernel(const int* __restrict__ x, int n, int* __restrict__ hist)
{
    const int lane = threadIdx.x & 31;
    const int nn   = lane & 15;
    const int half = lane >> 4;
    const int colorBase = half * 8;
    const int mBase = blockIdx.x * 16;
    const int m = mBase + nn;

    v8i b;
#pragma unroll
    for (int v = 0; v < 8; ++v)
        b[v] = ((v & 1) == ((nn >> 2) & 1) && half == (nn >> 3)) ? (int)(1u << ((nn & 3) * 8)) : 0;

    v8i acc = {};

    if (ROWM) {
        const int4* __restrict__ xv = (const int4*)(x + (size_t)m * n);
        for (int t0 = 0; t0 < n; t0 += 16) {
            int4 q[4];
#pragma unroll
            for (int u = 0; u < 4; ++u) q[u] = __ldg(&xv[(t0 >> 2) + u]);
#pragma unroll
            for (int u = 0; u < 4; ++u) {
                unsigned long long o0 = onehot8(q[u].x, colorBase);
                unsigned long long o1 = onehot8(q[u].y, colorBase);
                unsigned long long o2 = onehot8(q[u].z, colorBase);
                unsigned long long o3 = onehot8(q[u].w, colorBase);
                v8i a;
                a[0] = (int)(unsigned)o0; a[1] = (int)(unsigned)(o0 >> 32);
                a[2] = (int)(unsigned)o1; a[3] = (int)(unsigned)(o1 >> 32);
                a[4] = (int)(unsigned)o2; a[5] = (int)(unsigned)(o2 >> 32);
                a[6] = (int)(unsigned)o3; a[7] = (int)(unsigned)(o3 >> 32);
                acc = __builtin_amdgcn_wmma_i32_16x16x64_iu8(false, a, false, b,
                                                             acc, false, false);
            }
        }
    } else {
        for (int t0 = 0; t0 < n; t0 += 16) {
            int vals[16];
#pragma unroll
            for (int u = 0; u < 16; ++u)
                vals[u] = __ldg(&x[(size_t)(t0 + u) * n + m]);
#pragma unroll
            for (int u = 0; u < 4; ++u) {
                unsigned long long o0 = onehot8(vals[4 * u + 0], colorBase);
                unsigned long long o1 = onehot8(vals[4 * u + 1], colorBase);
                unsigned long long o2 = onehot8(vals[4 * u + 2], colorBase);
                unsigned long long o3 = onehot8(vals[4 * u + 3], colorBase);
                v8i a;
                a[0] = (int)(unsigned)o0; a[1] = (int)(unsigned)(o0 >> 32);
                a[2] = (int)(unsigned)o1; a[3] = (int)(unsigned)(o1 >> 32);
                a[4] = (int)(unsigned)o2; a[5] = (int)(unsigned)(o2 >> 32);
                a[6] = (int)(unsigned)o3; a[7] = (int)(unsigned)(o3 >> 32);
                acc = __builtin_amdgcn_wmma_i32_16x16x64_iu8(false, a, false, b,
                                                             acc, false, false);
            }
        }
    }

#pragma unroll
    for (int r = 0; r < 8; ++r) {
        int row = mBase + r + 8 * half;
        hist[row * NUM_COLORS + nn] = acc[r];
    }
}

__global__ void first_match_label_kernel(const int* __restrict__ hist, int n,
                                         int* __restrict__ lab)
{
    int i = blockIdx.x * blockDim.x + threadIdx.x;
    if (i >= n) return;
    int h[NUM_COLORS];
#pragma unroll
    for (int c = 0; c < NUM_COLORS; ++c) h[c] = hist[i * NUM_COLORS + c];
    int j = 0;
    for (; j < i; ++j) {
        bool eq = true;
#pragma unroll
        for (int c = 0; c < NUM_COLORS; ++c)
            eq = eq && (__ldg(&hist[j * NUM_COLORS + c]) == h[c]);
        if (eq) break;
    }
    lab[i] = j;
}

__device__ __forceinline__ unsigned mix32(unsigned k) {
    k ^= k >> 16; k *= 0x85ebca6bu;
    k ^= k >> 13; k *= 0xc2b2ae35u;
    k ^= k >> 16; return k;
}

__device__ __forceinline__ unsigned probe_find(const unsigned* __restrict__ keyTab,
                                               unsigned key, unsigned mask) {
    unsigned s = mix32(key) & mask;
    while (true) {
        if (keyTab[s] == key) return s;
        s = (s + 1) & mask;
    }
}

__global__ void fill_u32_kernel(unsigned* __restrict__ p, long long count, unsigned v) {
    for (long long i = blockIdx.x * (long long)blockDim.x + threadIdx.x;
         i < count; i += (long long)gridDim.x * blockDim.x)
        p[i] = v;
}

__device__ __forceinline__ unsigned cell_key(const int* __restrict__ x,
                                             const int* __restrict__ r,
                                             const int* __restrict__ c,
                                             int n, long long p) {
    int i = (int)(p / n), j = (int)(p % n);
    return ((unsigned)x[p] * (unsigned)n + (unsigned)r[i]) * (unsigned)n
           + (unsigned)c[j];
}

__global__ void insert_keys_kernel(const int* __restrict__ x,
                                   const int* __restrict__ r,
                                   const int* __restrict__ c, int n,
                                   unsigned* __restrict__ keyTab, unsigned mask)
{
    long long total = (long long)n * n;
    for (long long p = blockIdx.x * (long long)blockDim.x + threadIdx.x;
         p < total; p += (long long)gridDim.x * blockDim.x) {
        unsigned key = cell_key(x, r, c, n, p);
        unsigned s = mix32(key) & mask;
        while (true) {
            unsigned prev = atomicCAS(&keyTab[s], EMPTY_KEY, key);
            if (prev == EMPTY_KEY || prev == key) break;
            s = (s + 1) & mask;
        }
    }
}

__global__ void min_pos_kernel(const int* __restrict__ x,
                               const int* __restrict__ r,
                               const int* __restrict__ c, int n,
                               const unsigned* __restrict__ keyTab,
                               unsigned* __restrict__ posTab, unsigned mask)
{
    long long total = (long long)n * n;
    for (long long p = blockIdx.x * (long long)blockDim.x + threadIdx.x;
         p < total; p += (long long)gridDim.x * blockDim.x) {
        unsigned key = cell_key(x, r, c, n, p);
        unsigned s = probe_find(keyTab, key, mask);
        atomicMin(&posTab[s], (unsigned)p);
    }
}

__global__ void mark_first_kernel(const int* __restrict__ x,
                                  const int* __restrict__ r,
                                  const int* __restrict__ c, int n,
                                  const unsigned* __restrict__ keyTab,
                                  const unsigned* __restrict__ posTab,
                                  unsigned* __restrict__ rank, unsigned mask)
{
    long long total = (long long)n * n;
    for (long long p = blockIdx.x * (long long)blockDim.x + threadIdx.x;
         p < total; p += (long long)gridDim.x * blockDim.x) {
        unsigned key = cell_key(x, r, c, n, p);
        unsigned s = probe_find(keyTab, key, mask);
        rank[p] = (posTab[s] == (unsigned)p) ? 1u : 0u;
    }
}

__global__ __launch_bounds__(256)
void chunk_sums_kernel(const unsigned* __restrict__ rank, unsigned* __restrict__ sums)
{
    __shared__ unsigned lds[256];
    long long base = (long long)blockIdx.x * CHUNK + threadIdx.x * 16;
    unsigned s = 0;
#pragma unroll
    for (int k = 0; k < 16; ++k) s += rank[base + k];
    lds[threadIdx.x] = s;
    __syncthreads();
    for (int off = 128; off > 0; off >>= 1) {
        if ((int)threadIdx.x < off) lds[threadIdx.x] += lds[threadIdx.x + off];
        __syncthreads();
    }
    if (threadIdx.x == 0) sums[blockIdx.x] = lds[0];
}

__global__ void scan_sums_kernel(unsigned* __restrict__ sums, int m)
{
    unsigned run = 0;
    for (int i = 0; i < m; ++i) { unsigned v = sums[i]; sums[i] = run; run += v; }
}

__global__ __launch_bounds__(256)
void chunk_scan_kernel(unsigned* __restrict__ rank, const unsigned* __restrict__ sums)
{
    __shared__ unsigned lds[256];
    long long base = (long long)blockIdx.x * CHUNK + threadIdx.x * 16;
    unsigned v[16]; unsigned s = 0;
#pragma unroll
    for (int k = 0; k < 16; ++k) { v[k] = rank[base + k]; s += v[k]; }
    lds[threadIdx.x] = s;
    __syncthreads();
    for (int off = 1; off < 256; off <<= 1) {
        unsigned t = ((int)threadIdx.x >= off) ? lds[threadIdx.x - off] : 0u;
        __syncthreads();
        lds[threadIdx.x] += t;
        __syncthreads();
    }
    unsigned excl = (threadIdx.x ? lds[threadIdx.x - 1] : 0u) + sums[blockIdx.x];
#pragma unroll
    for (int k = 0; k < 16; ++k) { rank[base + k] = excl; excl += v[k]; }
}

__global__ void write_out_kernel(const int* __restrict__ x,
                                 const int* __restrict__ r,
                                 const int* __restrict__ c, int n,
                                 const unsigned* __restrict__ keyTab,
                                 const unsigned* __restrict__ posTab,
                                 const unsigned* __restrict__ rank,
                                 unsigned mask, float* __restrict__ out)
{
    long long total = (long long)n * n;
    for (long long p = blockIdx.x * (long long)blockDim.x + threadIdx.x;
         p < total; p += (long long)gridDim.x * blockDim.x) {
        unsigned key = cell_key(x, r, c, n, p);
        unsigned s  = probe_find(keyTab, key, mask);
        unsigned fp = posTab[s];
        out[p] = (float)rank[fp];
    }
}

extern "C" void kernel_launch(void* const* d_in, const int* in_sizes, int n_in,
                              void* d_out, int out_size, void* d_ws, size_t ws_size,
                              hipStream_t stream)
{
    const int* x = (const int*)d_in[0];
    long long total = (long long)in_sizes[0];
    int n = (int)llround(sqrt((double)total));
    float* out = (float*)d_out;

    const long long TAB = 1ll << 23;
    const unsigned mask = (unsigned)(TAB - 1);
    const int numChunks = (int)(total / CHUNK);

    char* ws = (char*)d_ws;
    size_t off = 0;
    auto carve = [&](size_t bytes) -> void* {
        void* p = ws + off;
        off = (off + bytes + 255) & ~(size_t)255;
        return p;
    };
    int*      histRows = (int*)     carve((size_t)n * NUM_COLORS * 4);
    int*      histCols = (int*)     carve((size_t)n * NUM_COLORS * 4);
    int*      rLab     = (int*)     carve((size_t)n * 4);
    int*      cLab     = (int*)     carve((size_t)n * 4);
    unsigned* keyTab   = (unsigned*)carve((size_t)TAB * 4);
    unsigned* posTab   = (unsigned*)carve((size_t)TAB * 4);
    unsigned* rank     = (unsigned*)carve((size_t)total * 4);
    unsigned* sums     = (unsigned*)carve((size_t)numChunks * 4);
    (void)ws_size; (void)n_in; (void)out_size;

    fill_u32_kernel<<<4096, 256, 0, stream>>>(keyTab, TAB, EMPTY_KEY);
    fill_u32_kernel<<<4096, 256, 0, stream>>>(posTab, TAB, 0xFFFFFFFFu);

    hist16_wmma_kernel<true ><<<n / 16, 32, 0, stream>>>(x, n, histRows);
    hist16_wmma_kernel<false><<<n / 16, 32, 0, stream>>>(x, n, histCols);

    first_match_label_kernel<<<(n + 127) / 128, 128, 0, stream>>>(histRows, n, rLab);
    first_match_label_kernel<<<(n + 127) / 128, 128, 0, stream>>>(histCols, n, cLab);

    insert_keys_kernel<<<4096, 256, 0, stream>>>(x, rLab, cLab, n, keyTab, mask);
    min_pos_kernel   <<<4096, 256, 0, stream>>>(x, rLab, cLab, n, keyTab, posTab, mask);
    mark_first_kernel<<<4096, 256, 0, stream>>>(x, rLab, cLab, n, keyTab, posTab, rank, mask);

    chunk_sums_kernel<<<numChunks, 256, 0, stream>>>(rank, sums);
    scan_sums_kernel <<<1, 1, 0, stream>>>(sums, numChunks);
    chunk_scan_kernel<<<numChunks, 256, 0, stream>>>(rank, sums);

    write_out_kernel<<<4096, 256, 0, stream>>>(x, rLab, cLab, n, keyTab, posTab,
                                               rank, mask, out);
}
